// HierarchicalReasoningModel_ACTV1Block_58643483460270
// MI455X (gfx1250) — compile-verified, matrix operand order checked by reading, not run on hardware
//
#include <hip/hip_runtime.h>
#include <hip/hip_bf16.h>

// ---------------------------------------------------------------------------
// CDNA5 WMMA (wave32): bf16 A/B fragments (16 bf16/lane = 8 VGPRs), f32 C/D
// accumulator (8 f32/lane). All LDS operand buffers use a *fragment-packed*
// layout  buf[(frag*32 + lane)*16 + h]  so a fragment load is one aligned
// 32-byte vector load (2x ds_load_b128) instead of 16x ds_load_u16.
//
// Element mapping inside a 16x32 (A) / 32x16 (B) fragment:
//   lane = (m_or_n & 15) | (((k >> 3) & 1) << 4)
//   h    = (k & 7)       | (((k >> 4) & 1) << 3)
// so for a fixed (lane, h-half) the 8 K-values are CONTIGUOUS -> staging can
// write contiguous 16-byte runs into LDS.
// ---------------------------------------------------------------------------
typedef __attribute__((ext_vector_type(16))) __bf16 bf16x16;
typedef __attribute__((ext_vector_type(8)))  float  f32x8;

// Software RNE fallback (kept for reference; native cast is preferred).
static __device__ __forceinline__ __bf16 f2bf_sw(float f) {
  unsigned int u = __builtin_bit_cast(unsigned int, f);
  unsigned int r = (u + 0x7FFFu + ((u >> 16) & 1u)) >> 16;
  return __builtin_bit_cast(__bf16, (unsigned short)r);
}

// Native f32 -> bf16 conversion: gfx1250 has hardware bf16 converts, so this
// lowers to a single v_cvt (pairs may fuse into v_cvt_pk_bf16_f32).
static __device__ __forceinline__ __bf16 f2bf(float f) {
  return (__bf16)f;
}

// Packed pair conversion.
static __device__ __forceinline__ void f2bf2(float lo, float hi, __bf16* dst) {
#if __has_builtin(__builtin_amdgcn_cvt_pk_bf16_f32)
  auto v = __builtin_amdgcn_cvt_pk_bf16_f32(lo, hi);
  dst[0] = v[0];
  dst[1] = v[1];
#else
  dst[0] = f2bf(lo);
  dst[1] = f2bf(hi);
#endif
}

// Convert an 8-float contiguous run into 8 contiguous bf16 (vectorizable store).
static __device__ __forceinline__ void cvt_run8(const float* __restrict__ src,
                                                __bf16* __restrict__ dst) {
#pragma unroll
  for (int i = 0; i < 4; ++i) f2bf2(src[2 * i], src[2 * i + 1], dst + 2 * i);
}

static __device__ __forceinline__ int laneid() { return (int)(threadIdx.x & 31u); }

// One aligned 32-byte load of a whole fragment for this lane.
static __device__ __forceinline__ bf16x16 frag_ld(const __bf16* buf, int f) {
  return *reinterpret_cast<const bf16x16*>(buf + (((size_t)f << 5) + laneid()) * 16);
}

// C/D 16x16 f32: VGPR r -> (M=r, N=lane) lanes 0-15 ; (M=r+8, N=lane-16) lanes 16-31.
static __device__ __forceinline__ void st_tile_c(float* base, int ld, const f32x8& c) {
  const int lane = laneid();
  const int n  = lane & 15;
  const int mo = (lane & 16) ? 8 : 0;
#pragma unroll
  for (int r = 0; r < 8; ++r) base[(r + mo) * ld + n] = c[r];
}

static __device__ __forceinline__ f32x8 wmma_bf16(bf16x16 a, bf16x16 b, f32x8 c) {
  return __builtin_amdgcn_wmma_f32_16x16x32_bf16(false, a, false, b, (short)0, c, false, false);
}

// Scatter-write helper (for the few producers that can't write contiguously):
// element (row_or_col index rc in 0..15 tile-local, k in 0..31) of fragment f.
static __device__ __forceinline__ int frag_pos(int f, int rc, int k) {
  const int lane = (rc & 15) | (((k >> 3) & 1) << 4);
  const int h    = (k & 7)   | (((k >> 4) & 1) << 3);
  return ((f << 5) + lane) * 16 + h;
}

// ---------------------------------------------------------------------------
// Problem constants
// ---------------------------------------------------------------------------
constexpr int T  = 4096;     // B*S
constexpr int S  = 1024;
constexpr int H  = 512;
constexpr int NH = 8;
constexpr int HD = 64;
constexpr int E  = 64;
constexpr int I  = 1536;
constexpr int GU_STRIDE = H * 2 * I;   // 512*3072
constexpr int D_STRIDE  = I * H;       // 1536*512

// ---------------------------------------------------------------------------
// Generic tiled GEMM: C[M,N] = A[M,K] @ B[K,N], f32 in/out, bf16 WMMA inside.
// 256 threads (8 waves), tile 64x64, K-step 32, fragment-packed LDS.
// ---------------------------------------------------------------------------
__global__ __launch_bounds__(256) void gemm_kernel(const float* __restrict__ A,
                                                   const float* __restrict__ B,
                                                   float* __restrict__ C,
                                                   int M, int N, int K) {
  __shared__ __bf16 As[4 * 32 * 16];   // 4 A-fragments (mi)
  __shared__ __bf16 Bs[4 * 32 * 16];   // 4 B-fragments (ni)
  const int bm = blockIdx.x * 64;
  const int bn = blockIdx.y * 64;
  const int tid = threadIdx.x;
  const int w = tid >> 5;
  const int mi = w & 3;
  const int nb = (w >> 2) * 2;

  // staging role: one A half-frag-lane + one B half-frag-lane per thread
  const int fl   = tid >> 1;       // frag-lane 0..127
  const int half = tid & 1;        // 0: h0-7 (k 0..7[+8g]) ; 1: h8-15 (k 16..23[+8g])
  const int f    = fl >> 5;        // fragment 0..3
  const int lane = fl & 31;
  const int g8   = (lane >> 4) * 8;
  const int kb   = (half ? 16 : 0) + g8;   // k-run base within 32

  f32x8 acc0 = {};
  f32x8 acc1 = {};

  for (int k0 = 0; k0 < K; k0 += 32) {
    {
      const int m = bm + f * 16 + (lane & 15);
      const float* src = A + (size_t)m * K + k0 + kb;
      __builtin_prefetch(src + 32, 0, 0);            // global_prefetch_b8
      cvt_run8(src, &As[fl * 16 + half * 8]);
    }
    {
      const int n = bn + f * 16 + (lane & 15);
      float v[8];
#pragma unroll
      for (int i = 0; i < 8; ++i) v[i] = B[(size_t)(k0 + kb + i) * N + n];
      cvt_run8(v, &Bs[fl * 16 + half * 8]);
    }
    __syncthreads();
    bf16x16 a  = frag_ld(As, mi);
    bf16x16 b0 = frag_ld(Bs, nb);
    bf16x16 b1 = frag_ld(Bs, nb + 1);
    acc0 = wmma_bf16(a, b0, acc0);
    acc1 = wmma_bf16(a, b1, acc1);
    __syncthreads();
  }
  st_tile_c(&C[(size_t)(bm + mi * 16) * N + bn + nb * 16], N, acc0);
  st_tile_c(&C[(size_t)(bm + mi * 16) * N + bn + (nb + 1) * 16], N, acc1);
}

// ---------------------------------------------------------------------------
// RoPE applied in place to q (cols 0..511) and k (cols 512..1023) of qkv.
// ---------------------------------------------------------------------------
__global__ __launch_bounds__(512) void rope_kernel(float* __restrict__ qkv,
                                                   const float* __restrict__ cosb,
                                                   const float* __restrict__ sinb) {
  const int t = blockIdx.x;
  const int tid = threadIdx.x;
  const int hd = tid >> 5;         // 16 heads (8 q + 8 k)
  const int d  = tid & 31;
  const int s  = t & (S - 1);
  const float c1 = cosb[s * HD + d],      s1 = sinb[s * HD + d];
  const float c2 = cosb[s * HD + d + 32], s2 = sinb[s * HD + d + 32];
  float* p = qkv + (size_t)t * (3 * H) + hd * HD + d;
  const float x1 = p[0], x2 = p[32];
  p[0]  = x1 * c1 - x2 * s1;
  p[32] = x2 * c2 + x1 * s2;
}

// ---------------------------------------------------------------------------
// Flash attention: grid (B*NH, S/64), 128 threads (4 waves).
// Q/K^T/V/P all fragment-packed in LDS; WMMA for Q*K^T and P*V.
// ---------------------------------------------------------------------------
__global__ __launch_bounds__(128) void flash_kernel(const float* __restrict__ qkv,
                                                    float* __restrict__ ctx) {
  const int bh = blockIdx.x;
  const int b = bh >> 3, h = bh & 7;
  const int qt = blockIdx.y;
  const int tid = threadIdx.x;
  const int w = tid >> 5;

  __shared__ __bf16 Qfr[8 * 32 * 16];  // A frags: f = mi*2 + kk   (rows x d)
  __shared__ __bf16 Kfr[8 * 32 * 16];  // B frags: f = kk*4 + ni   (d x kcol)
  __shared__ __bf16 Vfr[8 * 32 * 16];  // B frags: f = kk*4 + ni   (krow x d)
  __shared__ __bf16 Pfr[8 * 32 * 16];  // A frags: f = mi*2 + kk   (rows x kcol)
  __shared__ float  Sb[64 * 64];
  __shared__ float  mrow[64], lrow[64], arow[64];

  // ---- stage Q (contiguous 8-float runs from global, contiguous LDS stores) ----
  for (int fl = tid; fl < 256; fl += 128) {
    const int f = fl >> 5, lane = fl & 31;
    const int mi = f >> 1, kk = f & 1;
    const int qrow = b * S + qt * 64 + mi * 16 + (lane & 15);
    const int g8 = (lane >> 4) * 8;
#pragma unroll
    for (int half = 0; half < 2; ++half) {
      const int db = kk * 32 + (half ? 16 : 0) + g8;
      cvt_run8(qkv + (size_t)qrow * (3 * H) + h * HD + db, &Qfr[fl * 16 + half * 8]);
    }
  }
  if (tid < 64) { mrow[tid] = -3e38f; lrow[tid] = 0.f; }
  f32x8 o0 = {}, o1 = {}, o2 = {}, o3 = {};
  __syncthreads();

  const int lane = tid & 31;
  const int nn  = lane & 15;
  const int mo  = (lane & 16) ? 8 : 0;

  for (int kt = 0; kt < S / 64; ++kt) {
    // ---- stage K^T and V fragment-packed ----
    for (int fl = tid; fl < 256; fl += 128) {
      const int f = fl >> 5, ln = fl & 31;
      const int kk = f >> 2, ni = f & 3;
      const int g8 = (ln >> 4) * 8;
      {  // K^T: contraction dim = d ; n = key column (token)
        const int tok = b * S + kt * 64 + ni * 16 + (ln & 15);
        const float* src = qkv + (size_t)tok * (3 * H) + H + h * HD;
#pragma unroll
        for (int half = 0; half < 2; ++half) {
          const int db = kk * 32 + (half ? 16 : 0) + g8;
          cvt_run8(src + db, &Kfr[fl * 16 + half * 8]);
        }
      }
      {  // V: contraction dim = key token ; n = d
        const int d = ni * 16 + (ln & 15);
#pragma unroll
        for (int half = 0; half < 2; ++half) {
          const int krun = kk * 32 + (half ? 16 : 0) + g8;
          float v[8];
#pragma unroll
          for (int i = 0; i < 8; ++i)
            v[i] = qkv[(size_t)(b * S + kt * 64 + krun + i) * (3 * H) + 2 * H + h * HD + d];
          cvt_run8(v, &Vfr[fl * 16 + half * 8]);
        }
      }
    }
    __syncthreads();

    // ---- scores = Q K^T * scale ----
    {
      f32x8 sc[4] = {{}, {}, {}, {}};
#pragma unroll
      for (int kk = 0; kk < 2; ++kk) {
        bf16x16 a = frag_ld(Qfr, w * 2 + kk);
#pragma unroll
        for (int ni = 0; ni < 4; ++ni)
          sc[ni] = wmma_bf16(a, frag_ld(Kfr, kk * 4 + ni), sc[ni]);
      }
#pragma unroll
      for (int ni = 0; ni < 4; ++ni)
#pragma unroll
        for (int r = 0; r < 8; ++r)
          Sb[(w * 16 + mo + r) * 64 + ni * 16 + nn] = sc[ni][r] * 0.125f;  // 1/sqrt(64)
    }
    __syncthreads();

    // ---- online softmax (one thread per q row); P written fragment-packed ----
    if (tid < 64) {
      const float mold = mrow[tid];
      float mnew = mold;
      for (int j = 0; j < 64; ++j) mnew = fmaxf(mnew, Sb[tid * 64 + j]);
      const float alpha = __expf(mold - mnew);
      float psum = 0.f;
      const int fbase = (tid >> 4) * 2;
      for (int j = 0; j < 64; ++j) {
        const float pv = __expf(Sb[tid * 64 + j] - mnew);
        Pfr[frag_pos(fbase + (j >> 5), tid & 15, j & 31)] = f2bf(pv);
        psum += pv;
      }
      lrow[tid] = lrow[tid] * alpha + psum;
      mrow[tid] = mnew;
      arow[tid] = alpha;
    }
    __syncthreads();

#pragma unroll
    for (int r = 0; r < 8; ++r) {
      const float al = arow[w * 16 + mo + r];
      o0[r] *= al; o1[r] *= al; o2[r] *= al; o3[r] *= al;
    }
#pragma unroll
    for (int kk = 0; kk < 2; ++kk) {
      bf16x16 a = frag_ld(Pfr, w * 2 + kk);
      o0 = wmma_bf16(a, frag_ld(Vfr, kk * 4 + 0), o0);
      o1 = wmma_bf16(a, frag_ld(Vfr, kk * 4 + 1), o1);
      o2 = wmma_bf16(a, frag_ld(Vfr, kk * 4 + 2), o2);
      o3 = wmma_bf16(a, frag_ld(Vfr, kk * 4 + 3), o3);
    }
    __syncthreads();
  }

#pragma unroll
  for (int r = 0; r < 8; ++r) {
    const int row = w * 16 + mo + r;
    const float inv = 1.0f / lrow[row];
    const size_t orow = (size_t)(b * S + qt * 64 + row) * H + h * HD;
    ctx[orow +  0 + nn] = o0[r] * inv;
    ctx[orow + 16 + nn] = o1[r] * inv;
    ctx[orow + 32 + nn] = o2[r] * inv;
    ctx[orow + 48 + nn] = o3[r] * inv;
  }
}

// ---------------------------------------------------------------------------
// out = rms_norm(a + b) over last dim 512. One block (256 thr) per token.
// ---------------------------------------------------------------------------
__global__ __launch_bounds__(256) void add_rmsnorm_kernel(const float* __restrict__ a,
                                                          const float* __restrict__ b,
                                                          float* __restrict__ out) {
  const int t = blockIdx.x;
  const int tid = threadIdx.x;
  __shared__ float red[256];
  const size_t o = (size_t)t * H;
  const float v0 = a[o + tid] + b[o + tid];
  const float v1 = a[o + 256 + tid] + b[o + 256 + tid];
  red[tid] = v0 * v0 + v1 * v1;
  __syncthreads();
  for (int s = 128; s > 0; s >>= 1) {
    if (tid < s) red[tid] += red[tid + s];
    __syncthreads();
  }
  const float scale = rsqrtf(red[0] / (float)H + 1e-5f);
  out[o + tid] = v0 * scale;
  out[o + 256 + tid] = v1 * scale;
}

// ---------------------------------------------------------------------------
// Per-token routing: softmax(64), device top-3, masked top-6, softmax weights.
// ---------------------------------------------------------------------------
__global__ __launch_bounds__(64) void router_kernel(const float* __restrict__ logits,
                                                    float* __restrict__ Psum,
                                                    int* __restrict__ idxb,
                                                    float* __restrict__ wbuf,
                                                    int* __restrict__ cnt,
                                                    int* __restrict__ devcnt) {
  const int t = blockIdx.x;
  const int tid = threadIdx.x;
  __shared__ float pr[64], red[64];
  __shared__ int   seldev[3];
  __shared__ float wsel[6];
  __shared__ int   isel[6];

  const float lg = logits[(size_t)t * E + tid];
  red[tid] = lg; __syncthreads();
  for (int s = 32; s > 0; s >>= 1) { if (tid < s) red[tid] = fmaxf(red[tid], red[tid + s]); __syncthreads(); }
  const float m = red[0]; __syncthreads();
  const float ex = __expf(lg - m);
  red[tid] = ex; __syncthreads();
  for (int s = 32; s > 0; s >>= 1) { if (tid < s) red[tid] += red[tid + s]; __syncthreads(); }
  const float p = ex / red[0];
  pr[tid] = p;
  atomicAdd(&Psum[tid], p);
  __syncthreads();

  if (tid < 8) { float ss = 0; for (int j = 0; j < 8; ++j) ss += pr[tid * 8 + j]; red[tid] = ss; }
  __syncthreads();
  if (tid == 0) {
    float d[8];
    for (int i = 0; i < 8; ++i) d[i] = red[i];
    for (int k = 0; k < 3; ++k) {
      int bi = 0; float bv = -1e30f;
      for (int i = 0; i < 8; ++i) if (d[i] > bv) { bv = d[i]; bi = i; }
      seldev[k] = bi; d[bi] = -1e30f;
    }
  }
  __syncthreads();
  const int dev = tid >> 3;
  const bool ok = (dev == seldev[0]) || (dev == seldev[1]) || (dev == seldev[2]);
  red[tid] = ok ? p : -1e30f;
  __syncthreads();
  if (tid == 0) {
    for (int k = 0; k < 6; ++k) {
      int bi = 0; float bv = -1e30f;
      for (int i = 0; i < 64; ++i) if (red[i] > bv) { bv = red[i]; bi = i; }
      isel[k] = bi; wsel[k] = pr[bi]; red[bi] = -1e30f;
    }
    float mx = wsel[0];
    for (int k = 1; k < 6; ++k) mx = fmaxf(mx, wsel[k]);
    float ssum = 0;
    for (int k = 0; k < 6; ++k) { wsel[k] = __expf(wsel[k] - mx); ssum += wsel[k]; }
    for (int k = 0; k < 6; ++k) wsel[k] /= ssum;
  }
  __syncthreads();
  if (tid < 6) {
    idxb[t * 6 + tid] = isel[tid];
    wbuf[t * 6 + tid] = wsel[tid];
    atomicAdd(&cnt[isel[tid]], 1);
    atomicAdd(&devcnt[isel[tid] >> 3], 1);
  }
}

__global__ void scan_kernel(const int* __restrict__ cnt, int* __restrict__ offs,
                            int* __restrict__ fill) {
  if (threadIdx.x == 0) {
    int run = 0;
    for (int e = 0; e < E; ++e) { offs[e] = run; run += cnt[e]; }
  }
  if (threadIdx.x < E) fill[threadIdx.x] = 0;
}

__global__ __launch_bounds__(256) void scatter_kernel(const int* __restrict__ idxb,
                                                      const float* __restrict__ wbuf,
                                                      const int* __restrict__ offs,
                                                      int* __restrict__ fill,
                                                      int* __restrict__ etok,
                                                      float* __restrict__ ew) {
  const int t = blockIdx.x * 256 + threadIdx.x;
  if (t >= T) return;
  for (int k = 0; k < 6; ++k) {
    const int e = idxb[t * 6 + k];
    const int pos = atomicAdd(&fill[e], 1);
    const int slot = offs[e] + pos;
    etok[slot] = t;
    ew[slot] = wbuf[t * 6 + k];
  }
}

// ---------------------------------------------------------------------------
// Fused expert FFN, fragment-packed LDS everywhere. Per block = (expert,
// 64-token tile); 512 threads (16 waves), each wave owns a 64x32 output slice.
// 148 KB dynamic LDS (needs CDNA5's 320 KB WGP LDS).
// etok==nullptr => dense (shared-expert) mode, weight 1, identity token list.
// ---------------------------------------------------------------------------
__global__ __launch_bounds__(512) void expert_ffn_kernel(const float* __restrict__ X,
                                                         const float* __restrict__ WguBase,
                                                         const float* __restrict__ WdBase,
                                                         const int* __restrict__ etok,
                                                         const float* __restrict__ ew,
                                                         const int* __restrict__ offs,
                                                         const int* __restrict__ cnt,
                                                         float* __restrict__ Out) {
  extern __shared__ char smem[];
  __bf16* Xt   = (__bf16*)smem;                     // 64 A-frags: f = mi*16+kk (64 KB)
  __bf16* slab = (__bf16*)(smem + 65536);           // GU: 64 B-frags f=kk*4+ci (64 KB)
                                                    // D (reuse): 32 B-frags f=n/16 (32 KB)
  float*  Gb   = (float*)(smem + 131072);           // 64x32 f32 (8 KB)
  float*  Ub   = (float*)(smem + 131072 + 8192);    // 64x32 f32 (8 KB)
  __bf16* Ab   = (__bf16*)(smem + 131072 + 16384);  // 4 A-frags f = mi (4 KB)
  __shared__ int   tokr[64];
  __shared__ float wrow[64];

  const int tid = threadIdx.x;
  const int w = tid >> 5;
  int nvalid;
  const float* Wgu;
  const float* Wd;

  if (etok) {
    const int e = blockIdx.x >> 6;
    const int tile = blockIdx.x & 63;
    const int n = cnt[e];
    const int start = tile * 64;
    if (start >= n) return;
    nvalid = min(64, n - start);
    Wgu = WguBase + (size_t)e * GU_STRIDE;
    Wd  = WdBase  + (size_t)e * D_STRIDE;
    if (tid < 64) {
      if (tid < nvalid) {
        tokr[tid] = etok[offs[e] + start + tid];
        wrow[tid] = ew[offs[e] + start + tid];
      } else { tokr[tid] = -1; wrow[tid] = 0.f; }
    }
  } else {
    nvalid = 64;
    Wgu = WguBase; Wd = WdBase;
    if (tid < 64) { tokr[tid] = blockIdx.x * 64 + tid; wrow[tid] = 1.0f; }
  }
  __syncthreads();

  // ---- stage gathered X, fragment-packed (contiguous global + LDS runs) ----
  for (int fl = tid; fl < 2048; fl += 512) {
    const int f = fl >> 5, ln = fl & 31;
    const int mi = f >> 4, kk = f & 15;
    const int row = mi * 16 + (ln & 15);
    const int t = tokr[row];
    const int g8 = (ln >> 4) * 8;
#pragma unroll
    for (int half = 0; half < 2; ++half) {
      const int kb = kk * 32 + (half ? 16 : 0) + g8;
      __bf16* dst = &Xt[fl * 16 + half * 8];
      if (t >= 0) {
        cvt_run8(X + (size_t)t * H + kb, dst);
      } else {
#pragma unroll
        for (int i = 0; i < 8; ++i) dst[i] = (__bf16)(0.0f);
      }
    }
  }

  f32x8 zero8 = {};
  f32x8 acc[4][2];
#pragma unroll
  for (int a = 0; a < 4; ++a) { acc[a][0] = zero8; acc[a][1] = zero8; }
  __syncthreads();

  for (int ic = 0; ic < I / 32; ++ic) {
    // ---- stage gate|up B-fragments: f = kk*4 + ci, ci 0-1 gate / 2-3 up ----
    for (int fl = tid; fl < 2048; fl += 512) {
      const int f = fl >> 5, ln = fl & 31;
      const int kk = f >> 2, ci = f & 3;
      const int c = ci * 16 + (ln & 15);                 // 0..63 within chunk
      const int gcol = (c < 32) ? (ic * 32 + c) : (I + ic * 32 + (c - 32));
      const int g8 = (ln >> 4) * 8;
      __bf16* dst = &slab[fl * 16];
#pragma unroll
      for (int half = 0; half < 2; ++half) {
        const int kb = kk * 32 + (half ? 16 : 0) + g8;
        float v[8];
#pragma unroll
        for (int i = 0; i < 8; ++i) v[i] = Wgu[(size_t)(kb + i) * (2 * I) + gcol];
        cvt_run8(v, dst + half * 8);
      }
    }
    __syncthreads();

    // ---- gate/up: 16 tiles (8 gate + 8 up), one per wave, K=512 ----
    {
      const bool isU = w >= 8;
      const int li = isU ? (w - 8) : w;
      const int mi = li >> 1, ni = li & 1;
      const int ci = (isU ? 2 : 0) + ni;
      f32x8 s = zero8;
#pragma unroll
      for (int kk = 0; kk < 16; ++kk) {
        bf16x16 a  = frag_ld(Xt, mi * 16 + kk);
        bf16x16 bb = frag_ld(slab, kk * 4 + ci);
        s = wmma_bf16(a, bb, s);
      }
      st_tile_c((isU ? Ub : Gb) + (mi * 16) * 32 + ni * 16, 32, s);
    }
    __syncthreads();

    // ---- act = silu(gate) * up, written fragment-packed (A frags f = mi) ----
    for (int i = tid; i < 2048; i += 512) {
      const int row = i >> 5, col = i & 31;
      const float g = Gb[i], u = Ub[i];
      Ab[frag_pos(row >> 4, row & 15, col)] = f2bf((g / (1.0f + __expf(-g))) * u);
    }
    __syncthreads();

    // ---- stage down-proj B-fragments: 32 frags over n (K=32 single step) ----
    for (int fl = tid; fl < 1024; fl += 512) {
      const int f = fl >> 5, ln = fl & 31;
      const int n = f * 16 + (ln & 15);
      const int g8 = (ln >> 4) * 8;
      __bf16* dst = &slab[fl * 16];
#pragma unroll
      for (int half = 0; half < 2; ++half) {
        const int kb = (half ? 16 : 0) + g8;
        float v[8];
#pragma unroll
        for (int i = 0; i < 8; ++i) v[i] = Wd[(size_t)(ic * 32 + kb + i) * H + n];
        cvt_run8(v, dst + half * 8);
      }
    }
    __syncthreads();

    // ---- down: each wave its 64x32 slice ----
#pragma unroll
    for (int mi = 0; mi < 4; ++mi) {
      bf16x16 a = frag_ld(Ab, mi);
#pragma unroll
      for (int ni = 0; ni < 2; ++ni) {
        bf16x16 bb = frag_ld(slab, w * 2 + ni);
        acc[mi][ni] = wmma_bf16(a, bb, acc[mi][ni]);
      }
    }
    __syncthreads();
  }

  // ---- weighted scatter-add into accumulation buffer ----
  const int lane = tid & 31;
  const int n  = lane & 15;
  const int mo = (lane & 16) ? 8 : 0;
  const int n0 = w * 32;
#pragma unroll
  for (int mi = 0; mi < 4; ++mi)
#pragma unroll
    for (int ni = 0; ni < 2; ++ni)
#pragma unroll
      for (int r = 0; r < 8; ++r) {
        const int row = mi * 16 + mo + r;
        if (row < nvalid) {
          const int t = tokr[row];
          atomicAdd(&Out[(size_t)t * H + n0 + ni * 16 + n], wrow[row] * acc[mi][ni][r]);
        }
      }
}

// ---------------------------------------------------------------------------
// Balance losses.
// ---------------------------------------------------------------------------
__global__ void finalize_kernel(const int* __restrict__ cnt, const int* __restrict__ devcnt,
                                const float* __restrict__ Psum, float* __restrict__ out4) {
  if (threadIdx.x != 0) return;
  float f[E], P[E];
  for (int e = 0; e < E; ++e) {
    f[e] = (float)cnt[e] / ((float)T * 6.0f);
    P[e] = Psum[e] / (float)T;
  }
  float ebl = 0.f;
  for (int e = 0; e < E; ++e) ebl += f[e] * P[e];
  ebl *= 0.003f;
  float dbl = 0.f, cbl = 0.f;
  for (int d = 0; d < 8; ++d) {
    float df = 0.f, dP = 0.f;
    for (int j = 0; j < 8; ++j) { df += f[d * 8 + j]; dP += P[d * 8 + j]; }
    df /= 8.0f;
    dbl += df * dP;
    cbl += ((float)devcnt[d] / ((float)T * 3.0f)) * dP;
  }
  dbl *= 0.05f;
  cbl *= 0.02f;
  out4[0] = ebl; out4[1] = dbl; out4[2] = cbl; out4[3] = ebl + dbl + cbl;
}

// ---------------------------------------------------------------------------
// Host launch
// ---------------------------------------------------------------------------
extern "C" void kernel_launch(void* const* d_in, const int* in_sizes, int n_in,
                              void* d_out, int out_size, void* d_ws, size_t ws_size,
                              hipStream_t stream) {
  (void)in_sizes; (void)n_in; (void)out_size; (void)ws_size;

  const float* x        = (const float*)d_in[0];  // (4,1024,512)
  const float* cosb     = (const float*)d_in[1];  // (1024,64)
  const float* sinb     = (const float*)d_in[2];  // (1024,64)
  const float* w_qkv    = (const float*)d_in[3];  // (512,1536)
  const float* w_o      = (const float*)d_in[4];  // (512,512)
  const float* router_w = (const float*)d_in[5];  // (512,64)
  const float* r_wgu    = (const float*)d_in[6];  // (64,512,3072)
  const float* r_wd     = (const float*)d_in[7];  // (64,1536,512)
  const float* s_wgu    = (const float*)d_in[8];  // (2,512,3072)
  const float* s_wd     = (const float*)d_in[9];  // (2,1536,512)

  char* base = (char*)d_ws;
  size_t off = 0;
  auto carve = [&](size_t bytes) { char* p = base + off; off += (bytes + 255) & ~(size_t)255; return p; };

  float* qkvb   = (float*)carve((size_t)T * 3 * H * 4);  // 25 MB
  float* projb  = (float*)carve((size_t)T * H * 4);
  float* hbuf   = (float*)carve((size_t)T * H * 4);
  float* ctxb   = (float*)carve((size_t)T * H * 4);
  float* logits = (float*)carve((size_t)T * E * 4);
  float* accb   = (float*)carve((size_t)T * H * 4);      // shared + routed accumulation
  float* Psum   = (float*)carve(E * 4);
  float* wbuf   = (float*)carve((size_t)T * 6 * 4);
  float* ewb    = (float*)carve((size_t)T * 6 * 4);
  int*   idxb   = (int*)carve((size_t)T * 6 * 4);
  int*   etok   = (int*)carve((size_t)T * 6 * 4);
  int*   cntb   = (int*)carve(E * 4);
  int*   offsb  = (int*)carve(E * 4);
  int*   fillb  = (int*)carve(E * 4);
  int*   devcb  = (int*)carve(8 * 4);

  hipMemsetAsync(accb, 0, (size_t)T * H * 4, stream);
  hipMemsetAsync(Psum, 0, E * 4, stream);
  hipMemsetAsync(cntb, 0, E * 4, stream);
  hipMemsetAsync(devcb, 0, 8 * 4, stream);

  // Attention
  gemm_kernel<<<dim3(T / 64, (3 * H) / 64), 256, 0, stream>>>(x, w_qkv, qkvb, T, 3 * H, H);
  rope_kernel<<<T, 512, 0, stream>>>(qkvb, cosb, sinb);
  flash_kernel<<<dim3(4 * NH, S / 64), 128, 0, stream>>>(qkvb, ctxb);
  gemm_kernel<<<dim3(T / 64, H / 64), 256, 0, stream>>>(ctxb, w_o, projb, T, H, H);
  add_rmsnorm_kernel<<<T, 256, 0, stream>>>(x, projb, hbuf);

  // Routing
  gemm_kernel<<<dim3(T / 64, E / 64), 256, 0, stream>>>(hbuf, router_w, logits, T, E, H);
  router_kernel<<<T, 64, 0, stream>>>(logits, Psum, idxb, wbuf, cntb, devcb);
  scan_kernel<<<1, 64, 0, stream>>>(cntb, offsb, fillb);
  scatter_kernel<<<(T + 255) / 256, 256, 0, stream>>>(idxb, wbuf, offsb, fillb, etok, ewb);

  // Experts (routed sparse + 2 dense shared) — 148 KB dynamic LDS per block
  const int SMEM = 131072 + 16384 + 4096;  // 151552 bytes
  hipFuncSetAttribute(reinterpret_cast<const void*>(expert_ffn_kernel),
                      hipFuncAttributeMaxDynamicSharedMemorySize, SMEM);
  expert_ffn_kernel<<<E * 64, 512, SMEM, stream>>>(hbuf, r_wgu, r_wd, etok, ewb, offsb, cntb, accb);
  expert_ffn_kernel<<<T / 64, 512, SMEM, stream>>>(hbuf, s_wgu, s_wd,
                                                   nullptr, nullptr, nullptr, nullptr, accb);
  expert_ffn_kernel<<<T / 64, 512, SMEM, stream>>>(hbuf, s_wgu + (size_t)GU_STRIDE,
                                                   s_wd + (size_t)D_STRIDE,
                                                   nullptr, nullptr, nullptr, nullptr, accb);

  // Final norm + losses
  add_rmsnorm_kernel<<<T, 256, 0, stream>>>(hbuf, accb, (float*)d_out);
  finalize_kernel<<<1, 64, 0, stream>>>(cntb, devcb, Psum, (float*)d_out + (size_t)T * H);
}
